// MultiHeadAttentionBlock_38053410242965
// MI455X (gfx1250) — matrix operand order repaired and checked by reading, not compiled, not run on hardware
//
#include <hip/hip_runtime.h>

typedef __bf16 bf16_t;
typedef __attribute__((ext_vector_type(8)))  __bf16 v8bf;
typedef __attribute__((ext_vector_type(16))) __bf16 v16bf;
typedef __attribute__((ext_vector_type(8)))  float  v8f;

#define DEV __device__ __forceinline__

// ---- WMMA fragment helpers (16-bit operands, contiguous-k order used here, wave32) -------------------
// A operand (16x32 MxK): lane's row; components 0..15 = K kb..kb+15
// (contiguous), kb = (lane<16)?0:16.  Caller pre-offsets kb.
DEV v16bf load_a_frag(const bf16_t* p) {
  v8bf lo = *(const v8bf*)p;
  v8bf hi = *(const v8bf*)(p + 8);
  return __builtin_shufflevector(lo, hi, 0,1,2,3,4,5,6,7,8,9,10,11,12,13,14,15);
}
// B operand (32x16 KxN): lane's column; components 0..15 = K half*16..half*16+15
// (contiguous).  Caller pre-offsets half*16.
DEV v16bf load_b_frag(const bf16_t* p) {
  return *(const v16bf*)p;
}
DEV v8f wmma_bf16(v16bf a, v16bf b, v8f c) {
  return __builtin_amdgcn_wmma_f32_16x16x32_bf16(false, a, false, b, (short)0, c,
                                                 false, false);
}

// ---- fp32 -> bf16 convert ---------------------------------------------------
__global__ void __launch_bounds__(256)
f32_to_bf16(const float* __restrict__ in, bf16_t* __restrict__ out, int n) {
  int i = blockIdx.x * blockDim.x + threadIdx.x;
  if (i < n) out[i] = (bf16_t)in[i];
}

// ---- per-head V transpose: (BH,S,DK) -> (BH,DK,S) ---------------------------
__global__ void __launch_bounds__(256)
transpose_v(const bf16_t* __restrict__ V, bf16_t* __restrict__ Vt) {
  constexpr int S = 2048, DK = 64;
  size_t idx = (size_t)blockIdx.x * blockDim.x + threadIdx.x; // over BH*S*DK
  int dk = (int)(idx % DK);
  size_t r = idx / DK;
  int s = (int)(r % S);
  size_t bh = r / S;
  Vt[bh * (size_t)S * DK + (size_t)dk * S + s] = V[idx];
}

// ---- C = A(MxK) @ W(NxK)^T + bias, scaled; OUT = bf16 or f32 ---------------
// Wave computes a 32x64 tile (2 M-frags x 4 N-tiles, B frags reused);
// block = 8 waves -> 256x64 block tile.
template <typename OUT>
__global__ void __launch_bounds__(256)
gemm_xwT(const bf16_t* __restrict__ A, const bf16_t* __restrict__ W,
         const float* __restrict__ bias, OUT* __restrict__ out,
         int M, int N, int K, float scale) {
  const int lane = threadIdx.x & 31;
  const int wave = threadIdx.x >> 5;
  const int half = lane >> 4;
  const int lidx = lane & 15;
  const int m0 = blockIdx.x * 256 + wave * 32;
  const int n0 = blockIdx.y * 64;

  v8f acc[2][4];
#pragma unroll
  for (int mt = 0; mt < 2; ++mt)
#pragma unroll
    for (int t = 0; t < 4; ++t)
#pragma unroll
      for (int i = 0; i < 8; ++i) acc[mt][t][i] = 0.0f;

  const bf16_t* arow0 = A + (size_t)(m0 + lidx) * K + half * 16;
  const bf16_t* arow1 = arow0 + (size_t)16 * K;
  const bf16_t* wbase = W + (size_t)(n0 + lidx) * K + half * 16;

  for (int k0 = 0; k0 < K; k0 += 32) {
    __builtin_prefetch(arow0 + k0 + 128, 0, 1);
    __builtin_prefetch(arow1 + k0 + 128, 0, 1);
    v16bf af0 = load_a_frag(arow0 + k0);
    v16bf af1 = load_a_frag(arow1 + k0);
#pragma unroll
    for (int t = 0; t < 4; ++t) {
      v16bf bf = load_b_frag(wbase + (size_t)t * 16 * K + k0);
      acc[0][t] = wmma_bf16(af0, bf, acc[0][t]);
      acc[1][t] = wmma_bf16(af1, bf, acc[1][t]);
    }
  }
  // epilogue: C element (m0 + mt*16 + half*8 + v, n0 + t*16 + lidx)
#pragma unroll
  for (int mt = 0; mt < 2; ++mt)
#pragma unroll
    for (int t = 0; t < 4; ++t) {
      const int n = n0 + t * 16 + lidx;
      const float bv = bias[n];
#pragma unroll
      for (int v = 0; v < 8; ++v) {
        float r = (acc[mt][t][v] + bv) * scale;
        out[(size_t)(m0 + mt * 16 + half * 8 + v) * N + n] = (OUT)r;
      }
    }
}

// ---- fused causal attention per head (flash style) --------------------------
// Q,K flat (B,H,S,DK); Vt (B,H,DK,S); O flat (B,H,S,DK).  1 wave / 32 q-rows.
// Per 32-key step: 8 fragment loads feed 16 WMMAs (K/V frags reused over 2 m-tiles).
__global__ void __launch_bounds__(32)
flash_head(const bf16_t* __restrict__ Q, const bf16_t* __restrict__ Km,
           const bf16_t* __restrict__ Vt, bf16_t* __restrict__ O) {
  constexpr int S = 2048, DK = 64, H = 16;
  const int lane = threadIdx.x & 31;
  const int half = lane >> 4;
  const int lidx = lane & 15;
  const int kb = half * 16;
  const int q0 = blockIdx.x * 32;
  const size_t hoff = ((size_t)blockIdx.z * H + blockIdx.y) * (size_t)S * DK;
  const bf16_t* q  = Q  + hoff;
  const bf16_t* k  = Km + hoff;
  const bf16_t* vt = Vt + hoff;

  __shared__ bf16_t p_lds[2][16 * 32];

  // Q fragments: 2 m-tiles x dk chunks [0,32),[32,64); 1/sqrt(D) folded in.
  v16bf qf[2][2];
#pragma unroll
  for (int mt = 0; mt < 2; ++mt)
#pragma unroll
    for (int c = 0; c < 2; ++c)
      qf[mt][c] =
          load_a_frag(q + (size_t)(q0 + mt * 16 + lidx) * DK + c * 32 + kb);

  v8f acc[2][4];
  float mst[2][8], lst[2][8];
#pragma unroll
  for (int mt = 0; mt < 2; ++mt) {
#pragma unroll
    for (int t = 0; t < 4; ++t)
#pragma unroll
      for (int i = 0; i < 8; ++i) acc[mt][t][i] = 0.0f;
#pragma unroll
    for (int v = 0; v < 8; ++v) { mst[mt][v] = -1e30f; lst[mt][v] = 0.0f; }
  }

  const int nsteps = q0 / 32 + 1;  // causal: keys [0, 32*nsteps), j0 <= q0
  for (int step = 0; step < nsteps; ++step) {
    const int j0 = step * 32;
    // shared K fragments (2 j-tiles x 2 dk chunks) and V fragments (4 d-tiles)
    v16bf kf[2][2];
#pragma unroll
    for (int jt = 0; jt < 2; ++jt)
#pragma unroll
      for (int c = 0; c < 2; ++c)
        kf[jt][c] = load_b_frag(k + (size_t)(j0 + jt * 16 + lidx) * DK +
                                c * 32 + half * 16);
    v16bf vf[4];
#pragma unroll
    for (int t = 0; t < 4; ++t)
      vf[t] = load_b_frag(vt + (size_t)(t * 16 + lidx) * S + j0 + half * 16);

#pragma unroll
    for (int mt = 0; mt < 2; ++mt) {
      // scores for this m-tile: keys [j0,j0+16) and [j0+16,j0+32)
      v8f s0, s1;
#pragma unroll
      for (int i = 0; i < 8; ++i) { s0[i] = 0.0f; s1[i] = 0.0f; }
      s0 = wmma_bf16(qf[mt][0], kf[0][0], s0);
      s0 = wmma_bf16(qf[mt][1], kf[0][1], s0);
      s1 = wmma_bf16(qf[mt][0], kf[1][0], s1);
      s1 = wmma_bf16(qf[mt][1], kf[1][1], s1);

      // causal mask + per-row running max (rows: mt*16+half*8+v, cols: lidx)
      float tmax[8];
#pragma unroll
      for (int v = 0; v < 8; ++v) {
        const int irow = q0 + mt * 16 + half * 8 + v;
        float a0 = (j0 + lidx <= irow) ? s0[v] : -1e30f;
        float a1 = (j0 + 16 + lidx <= irow) ? s1[v] : -1e30f;
        s0[v] = a0; s1[v] = a1;
        tmax[v] = fmaxf(a0, a1);
      }
#pragma unroll
      for (int off = 8; off >= 1; off >>= 1)
#pragma unroll
        for (int v = 0; v < 8; ++v)
          tmax[v] = fmaxf(tmax[v], __shfl_xor(tmax[v], off, 32));
      float alpha[8], rsum[8];
#pragma unroll
      for (int v = 0; v < 8; ++v) {
        const float mnew = fmaxf(mst[mt][v], tmax[v]);
        alpha[v] = __expf(mst[mt][v] - mnew);
        const float p0 = __expf(s0[v] - mnew);
        const float p1 = __expf(s1[v] - mnew);
        mst[mt][v] = mnew;
        s0[v] = p0; s1[v] = p1;
        rsum[v] = p0 + p1;
      }
#pragma unroll
      for (int off = 8; off >= 1; off >>= 1)
#pragma unroll
        for (int v = 0; v < 8; ++v)
          rsum[v] += __shfl_xor(rsum[v], off, 32);
#pragma unroll
      for (int v = 0; v < 8; ++v)
        lst[mt][v] = lst[mt][v] * alpha[v] + rsum[v];
#pragma unroll
      for (int t = 0; t < 4; ++t)
#pragma unroll
        for (int v = 0; v < 8; ++v) acc[mt][t][v] *= alpha[v];

      // stage P (16 rows x 32 keys): C-layout -> A-layout via LDS
#pragma unroll
      for (int v = 0; v < 8; ++v) {
        p_lds[mt][(half * 8 + v) * 32 + lidx]      = (bf16_t)s0[v];
        p_lds[mt][(half * 8 + v) * 32 + 16 + lidx] = (bf16_t)s1[v];
      }
      // intra-wave LDS is in-order; fence keeps the scheduler honest.
      asm volatile("s_wait_dscnt 0x0" ::: "memory");
      v16bf pf = load_a_frag(&p_lds[mt][lidx * 32 + kb]);

      // acc += P @ V (Vt fragments shared across m-tiles)
#pragma unroll
      for (int t = 0; t < 4; ++t)
        acc[mt][t] = wmma_bf16(pf, vf[t], acc[mt][t]);
    }
  }

  // normalize + store bf16 (flat reshape-back is a no-op on this layout)
#pragma unroll
  for (int mt = 0; mt < 2; ++mt)
#pragma unroll
    for (int t = 0; t < 4; ++t)
#pragma unroll
      for (int v = 0; v < 8; ++v) {
        const int irow = q0 + mt * 16 + half * 8 + v;
        float r = acc[mt][t][v] / lst[mt][v];
        O[hoff + (size_t)irow * DK + t * 16 + lidx] = (bf16_t)r;
      }
}

extern "C" void kernel_launch(void* const* d_in, const int* in_sizes, int n_in,
                              void* d_out, int out_size, void* d_ws, size_t ws_size,
                              hipStream_t stream) {
  (void)in_sizes; (void)n_in; (void)out_size; (void)ws_size;
  constexpr int B = 2, S = 2048, D = 1024, H = 16;
  constexpr size_t NX = (size_t)B * S * D;  // 4,194,304
  constexpr size_t NW = (size_t)D * D;      // 1,048,576

  const float* key   = (const float*)d_in[0];
  const float* query = (const float*)d_in[1];
  const float* value = (const float*)d_in[2];
  // d_in[3] = mask (analytic causal; unused)
  const float* Wq = (const float*)d_in[4];
  const float* bq = (const float*)d_in[5];
  const float* Wk = (const float*)d_in[6];
  const float* bk = (const float*)d_in[7];
  const float* Wv = (const float*)d_in[8];
  const float* bv = (const float*)d_in[9];
  const float* Wo = (const float*)d_in[10];
  const float* bo = (const float*)d_in[11];

  char* ws = (char*)d_ws;
  bf16_t* xk  = (bf16_t*)ws; ws += NX * 2;
  bf16_t* xq  = (bf16_t*)ws; ws += NX * 2;
  bf16_t* xv  = (bf16_t*)ws; ws += NX * 2;
  bf16_t* wqb = (bf16_t*)ws; ws += NW * 2;
  bf16_t* wkb = (bf16_t*)ws; ws += NW * 2;
  bf16_t* wvb = (bf16_t*)ws; ws += NW * 2;
  bf16_t* wob = (bf16_t*)ws; ws += NW * 2;
  bf16_t* Qp  = (bf16_t*)ws; ws += NX * 2;
  bf16_t* Kp  = (bf16_t*)ws; ws += NX * 2;
  bf16_t* Vp  = (bf16_t*)ws; ws += NX * 2;
  bf16_t* Vtp = (bf16_t*)ws; ws += NX * 2;
  bf16_t* X2  = (bf16_t*)ws; ws += NX * 2;   // total 72 MB

  // 1) convert inputs + weights to bf16
  f32_to_bf16<<<dim3((unsigned)((NX + 255) / 256)), 256, 0, stream>>>(key,   xk, (int)NX);
  f32_to_bf16<<<dim3((unsigned)((NX + 255) / 256)), 256, 0, stream>>>(query, xq, (int)NX);
  f32_to_bf16<<<dim3((unsigned)((NX + 255) / 256)), 256, 0, stream>>>(value, xv, (int)NX);
  f32_to_bf16<<<dim3((unsigned)((NW + 255) / 256)), 256, 0, stream>>>(Wq, wqb, (int)NW);
  f32_to_bf16<<<dim3((unsigned)((NW + 255) / 256)), 256, 0, stream>>>(Wk, wkb, (int)NW);
  f32_to_bf16<<<dim3((unsigned)((NW + 255) / 256)), 256, 0, stream>>>(Wv, wvb, (int)NW);
  f32_to_bf16<<<dim3((unsigned)((NW + 255) / 256)), 256, 0, stream>>>(Wo, wob, (int)NW);

  // 2) projections (Q carries the 1/sqrt(D) = 1/32 score scale)
  const dim3 ggrd((B * S) / 256, D / 64);   // 16 x 16
  gemm_xwT<bf16_t><<<ggrd, 256, 0, stream>>>(xq, wqb, bq, Qp, B * S, D, D, 0.03125f);
  gemm_xwT<bf16_t><<<ggrd, 256, 0, stream>>>(xk, wkb, bk, Kp, B * S, D, D, 1.0f);
  gemm_xwT<bf16_t><<<ggrd, 256, 0, stream>>>(xv, wvb, bv, Vp, B * S, D, D, 1.0f);

  // 3) per-head V transpose for K-contiguous PV operand
  transpose_v<<<dim3((unsigned)(NX / 256)), 256, 0, stream>>>(Vp, Vtp);

  // 4) fused causal attention: 1 wave per 32 q-rows
  flash_head<<<dim3(S / 32, H, B), 32, 0, stream>>>(Qp, Kp, Vtp, X2);

  // 5) output projection, fp32 out
  gemm_xwT<float><<<ggrd, 256, 0, stream>>>(X2, wob, bo, (float*)d_out, B * S, D, D, 1.0f);
}
